// GroupedQueryAttention_33174327394329
// MI455X (gfx1250) — matrix operand order repaired and checked by reading, not compiled, not run on hardware
//
#include <hip/hip_runtime.h>

// ---------------------------------------------------------------- constants
constexpr int EMB  = 1024;
constexpr int NH   = 16;
constexpr int NKV  = 4;
constexpr int GS   = NH / NKV;   // 4 query heads per kv head
constexpr int HD   = EMB / NH;   // 64
constexpr int KVD  = NKV * HD;   // 256
constexpr int BB   = 2;
constexpr int SL   = 2048;
constexpr int MTOK = BB * SL;    // 4096 tokens

// ---------------------------------------------------------------- types
typedef __attribute__((ext_vector_type(16))) __bf16 v16bf;
typedef __attribute__((ext_vector_type(8)))  float  v8f;
typedef __attribute__((ext_vector_type(4)))  int    v4i;

// pointers into explicit address spaces for the async-LDS builtin
typedef __attribute__((address_space(1))) v4i* gv4i_p;   // global (__device__)
typedef __attribute__((address_space(3))) v4i* sv4i_p;   // LDS (__shared__)

union FragBF { v16bf v; uint4 u[2]; };

__device__ __forceinline__ v8f wmma_bf16(v16bf a, v16bf b, v8f c) {
  // (neg_a, A, neg_b, B, c_mod, C, reuse_a, reuse_b)
  return __builtin_amdgcn_wmma_f32_16x16x32_bf16(false, a, false, b,
                                                 (short)0, c, false, false);
}

__device__ __forceinline__ unsigned short f2bf(float f) {
  unsigned int u = __float_as_uint(f);
  u += 0x7FFFu + ((u >> 16) & 1u);      // round-to-nearest-even
  return (unsigned short)(u >> 16);
}

// CDNA5 async LDS fill (ASYNCcnt path) with safe fallback.
#if __has_builtin(__builtin_amdgcn_global_load_async_to_lds_b128) && \
    __has_builtin(__builtin_amdgcn_s_wait_asynccnt)
#define GQA_HAS_ASYNC_LDS 1
#else
#define GQA_HAS_ASYNC_LDS 0
#endif

// ---------------------------------------------------------------- precision convert
__global__ __launch_bounds__(256) void cvt_bf16_kernel(const float* __restrict__ src,
                                                       unsigned short* __restrict__ dst,
                                                       int n) {
  int i = blockIdx.x * blockDim.x + threadIdx.x;
  if (i < n) dst[i] = f2bf(src[i]);
}

// W [K x N] f32  ->  Wt [N x K] bf16 (so B-fragments are contiguous along K)
__global__ __launch_bounds__(256) void transpose_bf16_kernel(const float* __restrict__ W,
                                                             unsigned short* __restrict__ Wt,
                                                             int K, int N) {
  int i = blockIdx.x * blockDim.x + threadIdx.x;
  if (i < K * N) {
    int k = i / N, n = i % N;
    Wt[(size_t)n * K + k] = f2bf(W[i]);
  }
}

// ---------------------------------------------------------------- GEMM (bf16 x bf16 -> f32 acc)
// 8-wave workgroup computes a 128(M) x 64(N) block. B tile (64x32 bf16, 4 KB) is
// staged into a DOUBLE-BUFFERED LDS slab via async loads: while all 8 waves run
// WMMAs on buffer[cur], the next k-step's tile streams into buffer[cur^1]
// (s_wait_asynccnt(1) = previous load complete, next still in flight).
// Each wave reuses one A fragment across 4 WMMAs (16x64 per wave).
// EPI 0: store bf16 row-major [MxN]
// EPI 1: store bf16 into V-transposed layout [b, g, d, S]
// EPI 2: store f32 + bias into [MxN]
template <int M, int N, int K, int EPI>
__global__ __launch_bounds__(256)
void gemm_bf16_kernel(const unsigned short* __restrict__ A,
                      const unsigned short* __restrict__ Bt,
                      void* __restrict__ Cout,
                      const float* __restrict__ bias) {
  __shared__ __align__(16) unsigned short bTile[2][64 * 32];  // 2 x 4 KB staged B

  const int tid  = threadIdx.x;
  const int lane = tid & 31;
  const int wave = tid >> 5;
  const int l16  = lane & 15;
  const int hi   = lane >> 4;          // 0: lanes 0-15, 1: lanes 16-31

  constexpr int BLKN = N / 64;
  const int bn = (blockIdx.x % BLKN) * 64;    // block N origin
  const int bm = (blockIdx.x / BLKN) * 128;   // block M origin
  const int m0 = bm + wave * 16;              // this wave's M origin

  const unsigned short* arow = A + (size_t)(m0 + l16) * K;

  // cooperative B staging: thread -> (column n = tid>>2, 8-element chunk c = tid&3)
  const int sn = tid >> 2;                    // 0..63
  const int sc = tid & 3;                     // 0..3
  const unsigned short* bsrc = Bt + (size_t)(bn + sn) * K + sc * 8;
  const int sofs = sn * 32 + sc * 8;

  v8f acc[4] = {};

  // per-wave 16x64 compute step on one staged buffer
  auto compute_step = [&](const unsigned short* buf, int k0) {
    __builtin_prefetch(arow + k0 + 128, 0, 0);   // global_prefetch_b8
    FragBF a;
    // A 16x32 bf16 fragment: lane<16 -> K{0..7,16..23}, lane>=16 -> K{8..15,24..31}
    a.u[0] = *(const uint4*)(arow + k0 + hi * 8);
    a.u[1] = *(const uint4*)(arow + k0 + hi * 8 + 16);
#pragma unroll
    for (int j = 0; j < 4; ++j) {
      // B 32x16 fragment from LDS: lanes 0-15 K{0..7,16..23}, lanes 16-31 K{8..15,24..31}
      const unsigned short* bp = buf + (j * 16 + l16) * 32 + hi * 8;
      FragBF b;
      b.u[0] = *(const uint4*)(bp);
      b.u[1] = *(const uint4*)(bp + 16);
      acc[j] = wmma_bf16(a.v, b.v, acc[j]);
    }
  };

#if GQA_HAS_ASYNC_LDS
  // prologue: start filling buffer 0
  __builtin_amdgcn_global_load_async_to_lds_b128(
      (gv4i_p)(bsrc), (sv4i_p)(&bTile[0][sofs]), 0, 0);
  for (int k0 = 0; k0 < K; k0 += 32) {
    const int cur = (k0 >> 5) & 1;
    if (k0 + 32 < K) {
      // kick off next tile into the other buffer, then wait for current only
      __builtin_amdgcn_global_load_async_to_lds_b128(
          (gv4i_p)(bsrc + k0 + 32), (sv4i_p)(&bTile[cur ^ 1][sofs]), 0, 0);
      __builtin_amdgcn_s_wait_asynccnt(1);   // async loads complete in order
    } else {
      __builtin_amdgcn_s_wait_asynccnt(0);
    }
    __syncthreads();                         // all waves' stage-k0 data visible
    compute_step(&bTile[cur][0], k0);
    __syncthreads();                         // readers done before buffer reuse
  }
#else
  for (int k0 = 0; k0 < K; k0 += 32) {
    const int cur = (k0 >> 5) & 1;
    *(uint4*)(&bTile[cur][sofs]) = *(const uint4*)(bsrc + k0);
    __syncthreads();
    compute_step(&bTile[cur][0], k0);
    __syncthreads();
  }
#endif

  // C/D layout: VGPR r, lanes 0-15 -> (M=r, N=l16); lanes 16-31 -> (M=r+8, N=l16)
#pragma unroll
  for (int j = 0; j < 4; ++j) {
#pragma unroll
    for (int r = 0; r < 8; ++r) {
      const int row = m0 + r + hi * 8;
      const int col = bn + j * 16 + l16;
      if (EPI == 0) {
        ((unsigned short*)Cout)[(size_t)row * N + col] = f2bf(acc[j][r]);
      } else if (EPI == 1) {
        const int b2 = row / SL, s = row % SL;
        const int g = col >> 6, dc = col & 63;
        ((unsigned short*)Cout)[(((size_t)(b2 * NKV + g)) * HD + dc) * SL + s] =
            f2bf(acc[j][r]);
      } else {
        ((float*)Cout)[(size_t)row * N + col] = acc[j][r] + bias[col];
      }
    }
  }
}

// ---------------------------------------------------------------- flash attention
// One wave per block; block handles one (b, h, 16-row q tile). kv advances 32/iter.
__global__ __launch_bounds__(32)
void attn_kernel(const unsigned short* __restrict__ Qb,   // [MTOK, EMB] bf16
                 const unsigned short* __restrict__ Kb,   // [MTOK, KVD] bf16
                 const unsigned short* __restrict__ Vt,   // [b, g, d, S] bf16
                 unsigned short* __restrict__ ctx) {      // [MTOK, EMB] bf16
  __shared__ __align__(16) unsigned short pTile[16 * 32]; // P in C-layout -> A-layout bounce

  const int lane = threadIdx.x & 31;
  const int l16  = lane & 15;
  const int hi   = lane >> 4;

  const int QT = SL / 16;
  const int bh = blockIdx.x / QT;
  const int qt = blockIdx.x % QT;
  const int b2 = bh / NH;
  const int h  = bh % NH;
  const int g  = h / GS;
  const int q0 = qt * 16;

  // Q A-fragments for head-dim halves 0..31 / 32..63 (loaded once)
  const unsigned short* qrow = Qb + ((size_t)(b2 * SL) + q0 + l16) * EMB + h * HD;
  FragBF a0, a1;
  a0.u[0] = *(const uint4*)(qrow + hi * 8);
  a0.u[1] = *(const uint4*)(qrow + hi * 8 + 16);
  a1.u[0] = *(const uint4*)(qrow + hi * 8 + 32);
  a1.u[1] = *(const uint4*)(qrow + hi * 8 + 48);

  float m[8], l[8];
  v8f o[4] = {};
#pragma unroll
  for (int r = 0; r < 8; ++r) { m[r] = -1e30f; l[r] = 0.0f; }

  const float scale = 0.125f; // 1/sqrt(64)

  for (int kv0 = 0; kv0 < q0 + 16; kv0 += 32) {
    // ---- scores: two 16x16 tiles (cols kv0..+15 and kv0+16..+31), K=64 chained
    v8f s[2];
#pragma unroll
    for (int t = 0; t < 2; ++t) {
      const int kvcol = kv0 + t * 16 + l16;
      const unsigned short* krow =
          Kb + ((size_t)(b2 * SL) + kvcol) * KVD + g * HD + hi * 8;
      FragBF kb0, kb1;
      kb0.u[0] = *(const uint4*)(krow);
      kb0.u[1] = *(const uint4*)(krow + 16);
      kb1.u[0] = *(const uint4*)(krow + 32);
      kb1.u[1] = *(const uint4*)(krow + 48);
      v8f sc = {};
      sc = wmma_bf16(a0.v, kb0.v, sc);
      sc = wmma_bf16(a1.v, kb1.v, sc);
      s[t] = sc;
    }

    // ---- online softmax over the 32 new columns
#pragma unroll
    for (int r = 0; r < 8; ++r) {
      const int row = q0 + r + hi * 8;
      float v0 = s[0][r] * scale;
      float v1 = s[1][r] * scale;
      if (kv0 + l16 > row)      v0 = -1e30f;   // causal mask
      if (kv0 + 16 + l16 > row) v1 = -1e30f;

      float rmx = fmaxf(v0, v1);
      rmx = fmaxf(rmx, __shfl_xor(rmx, 1, 16));
      rmx = fmaxf(rmx, __shfl_xor(rmx, 2, 16));
      rmx = fmaxf(rmx, __shfl_xor(rmx, 4, 16));
      rmx = fmaxf(rmx, __shfl_xor(rmx, 8, 16));

      const float mnew = fmaxf(m[r], rmx);
      const float corr = __expf(m[r] - mnew);
      const float p0 = __expf(v0 - mnew);
      const float p1 = __expf(v1 - mnew);

      float rs = p0 + p1;
      rs += __shfl_xor(rs, 1, 16);
      rs += __shfl_xor(rs, 2, 16);
      rs += __shfl_xor(rs, 4, 16);
      rs += __shfl_xor(rs, 8, 16);

      l[r] = l[r] * corr + rs;
      m[r] = mnew;
      o[0][r] *= corr; o[1][r] *= corr; o[2][r] *= corr; o[3][r] *= corr;

      // scatter P (C-layout) into LDS as a row-major 16x32 bf16 tile
      const int prow = r + hi * 8;
      pTile[prow * 32 + l16]      = f2bf(p0);
      pTile[prow * 32 + 16 + l16] = f2bf(p1);
    }
    __syncthreads();  // single-wave WG: S_NOP, acts as compiler LDS fence

    // ---- gather P as a 16x32 A-fragment
    FragBF pA;
    const unsigned short* pl = &pTile[l16 * 32 + hi * 8];
    pA.u[0] = *(const uint4*)(pl);
    pA.u[1] = *(const uint4*)(pl + 16);

    // ---- O += P x V  (4 d-tiles of 16 columns; V pre-transposed, contiguous over kv)
#pragma unroll
    for (int dt = 0; dt < 4; ++dt) {
      const unsigned short* vcol =
          Vt + (((size_t)(b2 * NKV + g)) * HD + dt * 16 + l16) * SL + kv0 + hi * 8;
      FragBF vb;
      vb.u[0] = *(const uint4*)(vcol);
      vb.u[1] = *(const uint4*)(vcol + 16);
      o[dt] = wmma_bf16(pA.v, vb.v, o[dt]);
    }
    __syncthreads();
  }

  // ---- normalize and write ctx (bf16, token-major)
#pragma unroll
  for (int dt = 0; dt < 4; ++dt) {
#pragma unroll
    for (int r = 0; r < 8; ++r) {
      const int row = q0 + r + hi * 8;
      const float val = o[dt][r] * (1.0f / l[r]);
      ctx[((size_t)(b2 * SL) + row) * EMB + h * HD + dt * 16 + l16] = f2bf(val);
    }
  }
}

// ---------------------------------------------------------------- host launcher
extern "C" void kernel_launch(void* const* d_in, const int* in_sizes, int n_in,
                              void* d_out, int out_size, void* d_ws, size_t ws_size,
                              hipStream_t stream) {
  const float* x  = (const float*)d_in[0];
  const float* Wq = (const float*)d_in[1];
  const float* Wk = (const float*)d_in[2];
  const float* Wv = (const float*)d_in[3];
  const float* Wo = (const float*)d_in[4];
  const float* bo = (const float*)d_in[5];

  char* ws = (char*)d_ws;
  // workspace layout (bytes); total 33 MiB
  unsigned short* xb   = (unsigned short*)(ws + 0);                       // 8 MiB
  unsigned short* WqT  = (unsigned short*)(ws + 8388608);                 // 2 MiB
  unsigned short* WkT  = (unsigned short*)(ws + 10485760);                // 512 KiB
  unsigned short* WvT  = (unsigned short*)(ws + 11010048);                // 512 KiB
  unsigned short* WoT  = (unsigned short*)(ws + 11534336);                // 2 MiB
  unsigned short* Qb   = (unsigned short*)(ws + 13631488);                // 8 MiB
  unsigned short* Kb   = (unsigned short*)(ws + 22020096);                // 2 MiB
  unsigned short* Vt   = (unsigned short*)(ws + 24117248);                // 2 MiB
  unsigned short* ctxb = (unsigned short*)(ws + 26214400);                // 8 MiB

  // 1) precision conversion / weight transposes
  cvt_bf16_kernel<<<(MTOK * EMB) / 256, 256, 0, stream>>>(x, xb, MTOK * EMB);
  transpose_bf16_kernel<<<(EMB * EMB) / 256, 256, 0, stream>>>(Wq, WqT, EMB, EMB);
  transpose_bf16_kernel<<<(EMB * KVD) / 256, 256, 0, stream>>>(Wk, WkT, EMB, KVD);
  transpose_bf16_kernel<<<(EMB * KVD) / 256, 256, 0, stream>>>(Wv, WvT, EMB, KVD);
  transpose_bf16_kernel<<<(EMB * EMB) / 256, 256, 0, stream>>>(Wo, WoT, EMB, EMB);

  // 2) projections (WMMA bf16; 128x64 blocks, double-buffered async LDS B)
  gemm_bf16_kernel<MTOK, EMB, EMB, 0>
      <<<(MTOK / 128) * (EMB / 64), 256, 0, stream>>>(xb, WqT, Qb, nullptr);
  gemm_bf16_kernel<MTOK, KVD, EMB, 0>
      <<<(MTOK / 128) * (KVD / 64), 256, 0, stream>>>(xb, WkT, Kb, nullptr);
  gemm_bf16_kernel<MTOK, KVD, EMB, 1>
      <<<(MTOK / 128) * (KVD / 64), 256, 0, stream>>>(xb, WvT, Vt, nullptr);

  // 3) causal flash attention (WMMA bf16, online softmax)
  attn_kernel<<<BB * NH * (SL / 16), 32, 0, stream>>>(Qb, Kb, Vt, ctxb);

  // 4) output projection + bias -> f32 d_out
  gemm_bf16_kernel<MTOK, EMB, EMB, 2>
      <<<(MTOK / 128) * (EMB / 64), 256, 0, stream>>>(ctxb, WoT, d_out, bo);
}
